// MLSTM_721554505904
// MI455X (gfx1250) — matrix operand order repaired and checked by reading, not compiled, not run on hardware
//
#include <hip/hip_runtime.h>
#include <hip/hip_bf16.h>
#include <math.h>

#if __has_include(<hip/amd_detail/amd_gfx1250_TDM.h>)
#define TDM_SIX_ARGS 1
#else
#define TDM_SIX_ARGS 0
#endif

// ---------------------------------------------------------------------------
// CDNA5 (gfx1250) mLSTM: chunkwise-parallel scan + bf16 WMMA GEMMs.
// wave32 everywhere; matmuls -> v_wmma_f32_16x16x32_bf16; chunk staging -> TDM.
// ---------------------------------------------------------------------------

typedef __attribute__((ext_vector_type(16))) __bf16       v16bf;
typedef __attribute__((ext_vector_type(8)))  float        v8f;
typedef __attribute__((ext_vector_type(4)))  unsigned int u32x4;
typedef __attribute__((ext_vector_type(8)))  int          i32x8;
typedef __attribute__((ext_vector_type(4)))  int          i32x4;

#define WMMA_BF16(a, b, c) \
  __builtin_amdgcn_wmma_f32_16x16x32_bf16(false, (a), false, (b), (short)0, (c), false, false)

// A-fragment (16x32 MxK bf16): lane<16 row=lane K={0..7,16..23}; lane>=16 row=lane-16 K={8..15,24..31}
__device__ __forceinline__ int aKidx(int e, int lane) {
  return e + ((e & 8) ? 8 : 0) + ((lane & 16) ? 8 : 0);
}
// B-fragment (32x16 KxN bf16): N = lane%16; lanes<16 K={0..7,16..23}, lanes>=16 K={8..15,24..31}
__device__ __forceinline__ int bKidx(int e, int lane) {
  return e + ((e & 8) ? 8 : 0) + ((lane & 16) ? 8 : 0);
}
// C/D fragment (16x16 f32): VGPR r -> M = r + (lane>=16 ? 8 : 0), N = lane%16
__device__ __forceinline__ int cRow(int r, int lane) {
  return r + ((lane & 16) ? 8 : 0);
}

// ---------------------------------------------------------------------------
// Tensor Data Mover: contiguous 16KB (8192 x bf16) tile -> LDS with a 4-dword
// pad every 64 dwords (so LDS row pitch = 136 bf16, bank-conflict-free).
// ---------------------------------------------------------------------------
__device__ __forceinline__ void tdm_load_tile_bf16(const __bf16* gsrc, unsigned lds_byte_off) {
#if defined(__gfx1250__) && __has_builtin(__builtin_amdgcn_tensor_load_to_lds)
  const unsigned long long ga = (unsigned long long)(uintptr_t)gsrc;
  u32x4 g0 = (u32x4){1u /*count=1*/, lds_byte_off,
                     (unsigned)(ga & 0xffffffffull),
                     (unsigned)((ga >> 32) & 0x01ffffffull) | 0x80000000u /*type=2*/};
  // data_size=2B, pad_enable, pad_interval=64dw(code5), pad_amount=4dw(code3)
  // tensor_dim0=8192, tensor_dim1=1, tile_dim0=8192 (1D), dim0_stride=8192
  i32x8 g1 = (i32x8){0x07510000, 0x20000000, 0x00010000, 0x20000000,
                     0, 8192, 0, 0};
  i32x4 z4 = (i32x4){0, 0, 0, 0};
#if TDM_SIX_ARGS
  i32x8 z8 = (i32x8){0, 0, 0, 0, 0, 0, 0, 0};
  __builtin_amdgcn_tensor_load_to_lds(g0, g1, z4, z4, z8, 0);
#else
  __builtin_amdgcn_tensor_load_to_lds(g0, g1, z4, z4, 0);
#endif
#if __has_builtin(__builtin_amdgcn_s_wait_tensorcnt)
  __builtin_amdgcn_s_wait_tensorcnt(0);
#else
  asm volatile("s_wait_tensorcnt 0x0" ::: "memory");
#endif
#else
  (void)gsrc; (void)lds_byte_off;
#endif
}

__device__ __forceinline__ bool tdm_available() {
#if defined(__gfx1250__) && __has_builtin(__builtin_amdgcn_tensor_load_to_lds)
  return true;
#else
  return false;
#endif
}

// ---------------------------------------------------------------------------
// GEMM: Y[t,n] = act( alpha * sum_k X[t,k]*W[n,k] + bias[n] ),  N = 128.
// KDIM: 128 or 256 (k>=128 reads X2 -> concat([h,e]) FF).  W staged in LDS
// bf16 with +8 col pad.  256 thr = 8 waves, 16-row stripe/wave, 8 col tiles.
// ACT: 0 identity, 1 sigmoid, 2 leaky_relu(0.01).  OBF16: bf16 output buffer.
// ---------------------------------------------------------------------------
template <int KDIM, int ACT, bool OBF16>
__global__ __launch_bounds__(256) void gemm_xwt_kernel(
    const float* __restrict__ X1, const float* __restrict__ X2,
    const float* __restrict__ W,  const float* __restrict__ bias,
    void* __restrict__ Yv, float alpha)
{
  constexpr int WP = KDIM + 8;                  // padded LDS pitch (bf16)
  extern __shared__ __bf16 Wl[];                // [128 * WP]
  const int tid  = threadIdx.x;
  const int lane = tid & 31;
  const int wave = tid >> 5;

  for (int i = tid; i < 128 * KDIM; i += 256) {
    const int n = i / KDIM, k = i - n * KDIM;
    Wl[n * WP + k] = (__bf16)W[i];
  }
  __syncthreads();

  const int row0 = blockIdx.x * 128 + wave * 16;
  const int rA   = lane & 15;

  v8f acc[8];
#pragma unroll
  for (int j = 0; j < 8; ++j) acc[j] = (v8f){0.f,0.f,0.f,0.f,0.f,0.f,0.f,0.f};

#pragma unroll
  for (int ks = 0; ks < (KDIM >> 5); ++ks) {
    const int kb = ks << 5;
    const float* Xs = (kb < 128) ? X1 : X2;     // compile-time per unrolled ks
    const int    ko = (kb < 128) ? kb : kb - 128;
    v16bf a;
#pragma unroll
    for (int e = 0; e < 16; ++e)
      a[e] = (__bf16)Xs[(row0 + rA) * 128 + ko + aKidx(e, lane)];
#pragma unroll
    for (int nJ = 0; nJ < 8; ++nJ) {
      v16bf b;
#pragma unroll
      for (int e = 0; e < 16; ++e)
        b[e] = Wl[(nJ * 16 + rA) * WP + kb + bKidx(e, lane)];   // B[k,n] = W[n,k]
      acc[nJ] = WMMA_BF16(a, b, acc[nJ]);
    }
  }

  float*  Yf = (float*)Yv;
  __bf16* Yb = (__bf16*)Yv;
#pragma unroll
  for (int nJ = 0; nJ < 8; ++nJ) {
    const int col = nJ * 16 + rA;
    const float bv = bias[col];
#pragma unroll
    for (int r = 0; r < 8; ++r) {
      const int row = row0 + cRow(r, lane);
      float y = alpha * acc[nJ][r] + bv;
      if (ACT == 1) y = 1.f / (1.f + __expf(-y));
      if (ACT == 2) y = (y > 0.f) ? y : 0.01f * y;
      if (OBF16) Yb[row * 128 + col] = (__bf16)y;
      else       Yf[row * 128 + col] = y;
    }
  }
}

// ---------------------------------------------------------------------------
// Scalar gates: ii[t] = exp(x.w_i + b_i); logg[t] = log(sigmoid(x.w_f+b_f)*mask)
// clamped so exp() underflows to 0 across resets.  One wave32 per row.
// ---------------------------------------------------------------------------
__global__ __launch_bounds__(256) void gates_kernel(
    const float* __restrict__ X,
    const float* __restrict__ w_f, const float* __restrict__ b_f,
    const float* __restrict__ w_i, const float* __restrict__ b_i,
    const int*   __restrict__ start,
    float* __restrict__ ii, float* __restrict__ logg, int T)
{
  const int lane = threadIdx.x & 31;
  const int wave = threadIdx.x >> 5;
  const int t = blockIdx.x * (blockDim.x >> 5) + wave;
  if (t >= T) return;

  float ai = 0.f, af = 0.f;
#pragma unroll
  for (int p = 0; p < 4; ++p) {
    const float x = X[t * 128 + lane + 32 * p];
    ai += x * w_i[lane + 32 * p];
    af += x * w_f[lane + 32 * p];
  }
#pragma unroll
  for (int s = 16; s > 0; s >>= 1) {
    ai += __shfl_xor(ai, s, 32);
    af += __shfl_xor(af, s, 32);
  }
  if (lane == 0) {
    ii[t] = __expf(ai + b_i[0]);
    float g = 1.f / (1.f + __expf(-(af + b_f[0])));
    if (start[t]) g = 0.f;
    logg[t] = __logf(fmaxf(g, 1e-30f));
  }
}

// ---------------------------------------------------------------------------
// Chunkwise-parallel mLSTM scan. 512 thr = 16 waves, persistent; 32 chunks of
// c=64. LDS pitches padded for conflict-free fragment loads:
//   S  : 128 x 132 f32      Qc/Kc/Vc : 64 x 136 bf16 (TDM pad writes this!)
//   At : 64 x 72 bf16
// ---------------------------------------------------------------------------
#define SP 132
#define QP 136
#define AP 72

#define OFF_S   0u
#define OFF_Q   67584u
#define OFF_K   84992u
#define OFF_V   102400u
#define OFF_AT  119808u
#define OFF_M   129024u
#define OFF_BL  129536u
#define OFF_AL  129792u
#define OFF_II  130048u
#define OFF_WL  130304u
#define OFF_RS  130560u
#define OFF_DEN 130816u
#define OFF_AE  131072u
#define SCAN_SMEM_BYTES 131104u

__global__ __launch_bounds__(512) void mlstm_scan_kernel(
    const __bf16* __restrict__ Qg, const __bf16* __restrict__ Kg,
    const __bf16* __restrict__ Vg, const float* __restrict__ Og,
    const float* __restrict__ ii, const float* __restrict__ logg,
    float* __restrict__ H, int T)
{
  extern __shared__ char smem[];
  float*  S    = (float*)(smem + OFF_S);
  __bf16* Qc   = (__bf16*)(smem + OFF_Q);
  __bf16* Kc   = (__bf16*)(smem + OFF_K);
  __bf16* Vc   = (__bf16*)(smem + OFF_V);
  __bf16* At   = (__bf16*)(smem + OFF_AT);
  float*  m    = (float*)(smem + OFF_M);
  float*  bL   = (float*)(smem + OFF_BL);
  float*  aL   = (float*)(smem + OFF_AL);
  float*  iiL  = (float*)(smem + OFF_II);
  float*  wL   = (float*)(smem + OFF_WL);
  float*  rs   = (float*)(smem + OFF_RS);
  float*  den  = (float*)(smem + OFF_DEN);
  float*  aEnd = (float*)(smem + OFF_AE);

  const int tid  = threadIdx.x;
  const int lane = tid & 31;
  const int wave = tid >> 5;     // 16 waves
  const int rA   = lane & 15;

  for (int i = tid; i < 128 * SP; i += 512) S[i] = 0.f;
  if (tid < 128) m[tid] = 0.f;
  __syncthreads();

  const int NC = T >> 6;
  for (int ch = 0; ch < NC; ++ch) {
    const int t0 = ch << 6;

    // ---- stage chunk: TDM async tile loads (wave 0..2), or fallback copies ----
    if (tdm_available()) {
      if (wave == 0) tdm_load_tile_bf16(Qg + t0 * 128, OFF_Q);
      if (wave == 1) tdm_load_tile_bf16(Kg + t0 * 128, OFF_K);
      if (wave == 2) tdm_load_tile_bf16(Vg + t0 * 128, OFF_V);
    } else {
      for (int i = tid; i < 64 * 128; i += 512) {
        const int r = i >> 7, c = i & 127;
        Qc[r * QP + c] = Qg[t0 * 128 + i];
        Kc[r * QP + c] = Kg[t0 * 128 + i];
        Vc[r * QP + c] = Vg[t0 * 128 + i];
      }
    }
    if (ch + 1 < NC && tid * 16 < 64 * 128) {    // prefetch next chunk's o-gate
      __builtin_prefetch(&Og[(t0 + 64) * 128 + tid * 16], 0, 1);
    }
    if (tid < 64) rs[tid] = 0.f;
    __syncthreads();

    // ---- serial cumulative log-decay (one lane, 64 steps) ----
    if (tid == 0) {
      float b = 0.f;
      for (int j = 0; j < 64; ++j) {
        b += logg[t0 + j];
        bL[j]  = b;
        aL[j]  = __expf(b);
        iiL[j] = ii[t0 + j];
      }
      aEnd[0] = __expf(b);
      for (int j = 0; j < 64; ++j) wL[j] = __expf(b - bL[j]) * iiL[j];
    }
    __syncthreads();

    // ---- A = Q K^T, decay-mask -> At, row sums (ds_add_f32). 1 tile/wave ----
    {
      const int tI = wave >> 2, tJ = wave & 3;
      v8f acc = (v8f){0.f,0.f,0.f,0.f,0.f,0.f,0.f,0.f};
#pragma unroll
      for (int ks = 0; ks < 4; ++ks) {
        const int kb = ks << 5;
        v16bf a, b;
#pragma unroll
        for (int e = 0; e < 16; ++e)
          a[e] = Qc[(tI * 16 + rA) * QP + kb + aKidx(e, lane)];
#pragma unroll
        for (int e = 0; e < 16; ++e)
          b[e] = Kc[(tJ * 16 + rA) * QP + kb + bKidx(e, lane)];   // B[k',j]=K[j,k']
        acc = WMMA_BF16(a, b, acc);
      }
#pragma unroll
      for (int r = 0; r < 8; ++r) {
        const int tl = tI * 16 + cRow(r, lane);
        const int jl = tJ * 16 + rA;
        float av = 0.f;
        if (jl <= tl) av = __expf(bL[tl] - bL[jl]) * iiL[jl] * acc[r];
        At[tl * AP + jl] = (__bf16)av;
        if (av != 0.f) atomicAdd(&rs[tl], av);
      }
    }
    __syncthreads();

    // ---- denominators ----
    if (tid < 64) {
      float mq = 0.f;
      for (int k = 0; k < 128; ++k) mq += m[k] * (float)Qc[tid * QP + k];
      den[tid] = fmaxf(fabsf(aL[tid] * mq + rs[tid]), 1.f);
    }
    __syncthreads();

    // ---- num = At@V + a_t*(Q@S^T); H = o*num/den. 2 tiles/wave ----
    for (int p = 0; p < 2; ++p) {
      const int tile = wave * 2 + p;
      const int tI = tile >> 3, hJ = tile & 7;
      v8f nin = (v8f){0.f,0.f,0.f,0.f,0.f,0.f,0.f,0.f};
      v8f nxt = (v8f){0.f,0.f,0.f,0.f,0.f,0.f,0.f,0.f};
#pragma unroll
      for (int ks = 0; ks < 2; ++ks) {                     // intra: contract j
        const int kb = ks << 5;
        v16bf a, b;
#pragma unroll
        for (int e = 0; e < 16; ++e)
          a[e] = At[(tI * 16 + rA) * AP + kb + aKidx(e, lane)];
#pragma unroll
        for (int e = 0; e < 16; ++e)
          b[e] = Vc[(kb + bKidx(e, lane)) * QP + hJ * 16 + rA];
        nin = WMMA_BF16(a, b, nin);
      }
#pragma unroll
      for (int ks = 0; ks < 4; ++ks) {                     // inter: contract k'
        const int kb = ks << 5;
        v16bf a, b;
#pragma unroll
        for (int e = 0; e < 16; ++e)
          a[e] = Qc[(tI * 16 + rA) * QP + kb + aKidx(e, lane)];
#pragma unroll
        for (int e = 0; e < 16; ++e)
          b[e] = (__bf16)S[(hJ * 16 + rA) * SP + kb + bKidx(e, lane)]; // B[k',h]=S[h,k']
        nxt = WMMA_BF16(a, b, nxt);
      }
#pragma unroll
      for (int r = 0; r < 8; ++r) {
        const int tl = tI * 16 + cRow(r, lane);
        const int hc = hJ * 16 + rA;
        const float num = nin[r] + aL[tl] * nxt[r];
        const float oo  = Og[(t0 + tl) * 128 + hc];
        H[(t0 + tl) * 128 + hc] = oo * num / den[tl];
      }
    }
    __syncthreads();

    // ---- state update: S = a_end*S + (w.V)^T K. 4 tiles/wave ----
    {
      const float aE = aEnd[0];
      for (int p = 0; p < 4; ++p) {
        const int tile = wave * 4 + p;
        const int I = tile >> 3, J = tile & 7;
        v8f c;
#pragma unroll
        for (int r = 0; r < 8; ++r)
          c[r] = aE * S[(I * 16 + cRow(r, lane)) * SP + J * 16 + rA];
#pragma unroll
        for (int ks = 0; ks < 2; ++ks) {
          const int kb = ks << 5;
          v16bf a, b;
#pragma unroll
          for (int e = 0; e < 16; ++e) {                   // A[h,j] = w_j*V[j,h]
            const int j = kb + aKidx(e, lane);
            a[e] = (__bf16)(wL[j] * (float)Vc[j * QP + I * 16 + rA]);
          }
#pragma unroll
          for (int e = 0; e < 16; ++e) {                   // B[j,k'] = K[j,k']
            const int j = kb + bKidx(e, lane);
            b[e] = Kc[j * QP + J * 16 + rA];
          }
          c = WMMA_BF16(a, b, c);
        }
#pragma unroll
        for (int r = 0; r < 8; ++r)
          S[(I * 16 + cRow(r, lane)) * SP + J * 16 + rA] = c[r];
      }
      if (tid < 128) {
        float mv = aE * m[tid];
        for (int j = 0; j < 64; ++j) mv += wL[j] * (float)Kc[j * QP + tid];
        m[tid] = mv;
      }
    }
    __syncthreads();
  }
}

// ---------------------------------------------------------------------------
// Host-side orchestration
// ---------------------------------------------------------------------------
extern "C" void kernel_launch(void* const* d_in, const int* in_sizes, int n_in,
                              void* d_out, int out_size, void* d_ws, size_t ws_size,
                              hipStream_t stream) {
  (void)in_sizes; (void)n_in; (void)out_size; (void)ws_size;

  const int T = 2048;
  const float kscale = 0.0883883476483184f;   // 1/sqrt(128)

  const float* emb   = (const float*)d_in[0];
  const int*   start = (const int*)  d_in[1];
  const float* w_in  = (const float*)d_in[2];
  const float* b_in  = (const float*)d_in[3];
  const float* w_out = (const float*)d_in[4];
  const float* b_out = (const float*)d_in[5];
  const float* w_f   = (const float*)d_in[6];
  const float* b_f   = (const float*)d_in[7];
  const float* w_i   = (const float*)d_in[8];
  const float* b_i   = (const float*)d_in[9];
  const float* w_v   = (const float*)d_in[10];
  const float* b_v   = (const float*)d_in[11];
  const float* w_q   = (const float*)d_in[12];
  const float* b_q   = (const float*)d_in[13];
  const float* w_k   = (const float*)d_in[14];
  const float* b_k   = (const float*)d_in[15];
  const float* w_o   = (const float*)d_in[16];
  const float* b_o   = (const float*)d_in[17];
  const float* w_ff  = (const float*)d_in[18];
  const float* b_ff  = (const float*)d_in[19];

  float*  ws  = (float*)d_ws;
  float*  eB  = ws;                       // [T,128] f32
  float*  z1  = eB + T * 128;
  float*  z2  = z1 + T * 128;
  float*  oB  = z2 + T * 128;             // [T,128] f32
  float*  hB  = oB + T * 128;             // [T,128] f32
  float*  iiB = hB + T * 128;             // [T]
  float*  lgB = iiB + T;                  // [T]
  __bf16* qBb = (__bf16*)(lgB + T);       // [T,128] bf16
  __bf16* kBb = qBb + T * 128;
  __bf16* vBb = kBb + T * 128;

  {
    const void* f;
    f = reinterpret_cast<const void*>(&gemm_xwt_kernel<128, 0, false>);
    (void)hipFuncSetAttribute(f, hipFuncAttributeMaxDynamicSharedMemorySize, 34816);
    f = reinterpret_cast<const void*>(&gemm_xwt_kernel<128, 0, true>);
    (void)hipFuncSetAttribute(f, hipFuncAttributeMaxDynamicSharedMemorySize, 34816);
    f = reinterpret_cast<const void*>(&gemm_xwt_kernel<128, 1, false>);
    (void)hipFuncSetAttribute(f, hipFuncAttributeMaxDynamicSharedMemorySize, 34816);
    f = reinterpret_cast<const void*>(&gemm_xwt_kernel<256, 2, false>);
    (void)hipFuncSetAttribute(f, hipFuncAttributeMaxDynamicSharedMemorySize, 67584);
    f = reinterpret_cast<const void*>(&mlstm_scan_kernel);
    (void)hipFuncSetAttribute(f, hipFuncAttributeMaxDynamicSharedMemorySize,
                              (int)SCAN_SMEM_BYTES);
  }

  const dim3 gGrid(T / 128), gBlk(256);
  const unsigned sm128 = 128 * (128 + 8) * 2;   // 34816
  const unsigned sm256 = 128 * (256 + 8) * 2;   // 67584

  // e = emb @ w_in.T + b_in
  gemm_xwt_kernel<128, 0, false><<<gGrid, gBlk, sm128, stream>>>(
      emb, nullptr, w_in, b_in, eB, 1.f);

  for (int l = 0; l < 2; ++l) {
    const float* x  = l ? z1 : eB;
    float*       zo = l ? z2 : z1;

    gates_kernel<<<dim3(T / 8), gBlk, 0, stream>>>(
        x, w_f + l * 128, b_f + l, w_i + l * 128, b_i + l, start, iiB, lgB, T);

    gemm_xwt_kernel<128, 0, true><<<gGrid, gBlk, sm128, stream>>>(
        x, nullptr, w_v + l * 16384, b_v + l * 128, vBb, 1.f);
    gemm_xwt_kernel<128, 0, true><<<gGrid, gBlk, sm128, stream>>>(
        x, nullptr, w_k + l * 16384, b_k + l * 128, kBb, kscale);
    gemm_xwt_kernel<128, 0, true><<<gGrid, gBlk, sm128, stream>>>(
        x, nullptr, w_q + l * 16384, b_q + l * 128, qBb, 1.f);
    gemm_xwt_kernel<128, 1, false><<<gGrid, gBlk, sm128, stream>>>(
        x, nullptr, w_o + l * 16384, b_o + l * 128, oB, 1.f);

    mlstm_scan_kernel<<<dim3(1), dim3(512), SCAN_SMEM_BYTES, stream>>>(
        qBb, kBb, vBb, oB, iiB, lgB, hB, T);

    // z = leaky_relu(concat([h, e]) @ w_ff.T + b_ff)
    gemm_xwt_kernel<256, 2, false><<<gGrid, gBlk, sm256, stream>>>(
        hB, eB, w_ff + l * 128 * 256, b_ff + l * 128, zo, 1.f);
  }

  // out = z2 @ w_out.T + b_out
  gemm_xwt_kernel<128, 0, false><<<gGrid, gBlk, sm128, stream>>>(
      z2, nullptr, w_out, b_out, d_out, 1.f);
}
